// SortNet_30528627540612
// MI455X (gfx1250) — matrix operand order repaired and checked by reading, not compiled, not run on hardware
//
#include <hip/hip_runtime.h>
#include <stdint.h>

// ---------------------------------------------------------------------------
// MI455X (gfx1250) implementation of SortNet forward.
// GEMMs use v_wmma_f32_16x16x32_f16 (f16 in, f32 accum); fragment loads are
// branch-free vectorized b128 loads. FFN stages X tiles via the Tensor Data
// Mover (tensor_load_to_lds, 6-arg clang-23 form).
// ---------------------------------------------------------------------------

#define B_      8
#define N_      2048
#define F_      64
#define ROWS_   (B_ * N_)      // 16384
#define NHEAD_  8
#define HD_     8
#define TOPK_   16
#define NSAMP_  64
#define R2_     0.09f          // RADIUS^2
#define EPS_    1e-5f

typedef __attribute__((ext_vector_type(16))) _Float16 v16h;
typedef __attribute__((ext_vector_type(8)))  float    v8f;
typedef __attribute__((ext_vector_type(4)))  unsigned u32x4;
typedef __attribute__((ext_vector_type(8)))  int      i32x8;
typedef __attribute__((ext_vector_type(4)))  int      i32x4;

#define WMMA(a, b, c) \
  __builtin_amdgcn_wmma_f32_16x16x32_f16(false, (a), false, (b), (short)0, (c), false, false)

// ---------------- fast fragment loaders (no bounds, vectorized) ------------
// A fragment 16x32 f16: m = lane&15; k = (h&7) + 8*(lane>=16) + 16*(h>=8).
// Requires lda % 4 == 0 and k-base alignment (k0 % 32 == 0). Two b128 pairs.
__device__ __forceinline__ v16h frag_a_fast(const float* __restrict__ A, long lda,
                                            int m0, int k0, int lane) {
  const float* row = A + (long)(m0 + (lane & 15)) * lda + k0 + ((lane >> 4) << 3);
  float4 x0 = ((const float4*)row)[0];
  float4 x1 = ((const float4*)row)[1];
  float4 y0 = ((const float4*)(row + 16))[0];
  float4 y1 = ((const float4*)(row + 16))[1];
  v16h a;
  a[0]  = (_Float16)x0.x; a[1]  = (_Float16)x0.y;
  a[2]  = (_Float16)x0.z; a[3]  = (_Float16)x0.w;
  a[4]  = (_Float16)x1.x; a[5]  = (_Float16)x1.y;
  a[6]  = (_Float16)x1.z; a[7]  = (_Float16)x1.w;
  a[8]  = (_Float16)y0.x; a[9]  = (_Float16)y0.y;
  a[10] = (_Float16)y0.z; a[11] = (_Float16)y0.w;
  a[12] = (_Float16)y1.x; a[13] = (_Float16)y1.y;
  a[14] = (_Float16)y1.z; a[15] = (_Float16)y1.w;
  return a;
}

// B fragment 32x16 f16, B[k][n] = W[n][k]: n = lane&15; k = (h&7) + 8*(lane>=16) + 16*(h>=8).
// 8 floats at row[0..7] and 8 at row[16..23] per lane -> 4 b128 loads.
__device__ __forceinline__ v16h frag_b_fast(const float* __restrict__ W, long ldw,
                                            int n0, int k0, int lane) {
  const float* row = W + (long)(n0 + (lane & 15)) * ldw + k0 + ((lane >> 4) << 3);
  float4 x0 = ((const float4*)row)[0];
  float4 x1 = ((const float4*)row)[1];
  float4 x2 = ((const float4*)(row + 16))[0];
  float4 x3 = ((const float4*)(row + 16))[1];
  v16h b;
  b[0]  = (_Float16)x0.x; b[1]  = (_Float16)x0.y;
  b[2]  = (_Float16)x0.z; b[3]  = (_Float16)x0.w;
  b[4]  = (_Float16)x1.x; b[5]  = (_Float16)x1.y;
  b[6]  = (_Float16)x1.z; b[7]  = (_Float16)x1.w;
  b[8]  = (_Float16)x2.x; b[9]  = (_Float16)x2.y;
  b[10] = (_Float16)x2.z; b[11] = (_Float16)x2.w;
  b[12] = (_Float16)x3.x; b[13] = (_Float16)x3.y;
  b[14] = (_Float16)x3.z; b[15] = (_Float16)x3.w;
  return b;
}

// ---------------- edge fragment loaders (branchless clamp + select) --------
__device__ __forceinline__ v16h frag_a_edge(const float* __restrict__ A, long lda,
                                            int m0, int k0, int M, int K, int lane) {
  int m = m0 + (lane & 15);
  float mok = (m < M) ? 1.0f : 0.0f;
  m = min(m, M - 1);
  int kb = k0 + ((lane >> 4) << 3);
  const float* row = A + (long)m * lda;
  v16h a;
#pragma unroll
  for (int h = 0; h < 16; ++h) {
    int k = kb + (h & 7) + ((h >> 3) << 4);
    float v = row[min(k, K - 1)];      // unconditional load (clamped, in-bounds)
    v = (k < K) ? v : 0.0f;            // v_cndmask, no branch
    a[h] = (_Float16)(v * mok);
  }
  return a;
}

__device__ __forceinline__ v16h frag_b_edge(const float* __restrict__ W, long ldw,
                                            int n0, int k0, int N, int K, int lane) {
  int n = n0 + (lane & 15);
  float nok = (n < N) ? 1.0f : 0.0f;
  n = min(n, N - 1);
  int kb = k0 + ((lane >> 4) << 3);
  const float* row = W + (long)n * ldw;
  v16h b;
#pragma unroll
  for (int h = 0; h < 16; ++h) {
    int k = kb + (h & 7) + ((h >> 3) << 4);
    float v = row[min(k, K - 1)];
    v = (k < K) ? v : 0.0f;
    b[h] = (_Float16)(v * nok);
  }
  return b;
}

// ---------------- LDS fragment loaders -------------------------------------
__device__ __forceinline__ v16h frag_a_lds32(const float* X, int lda, int k0, int lane) {
  int m  = lane & 15;
  int kb = k0 + ((lane >> 4) << 3);
  const float* row = X + m * lda;
  v16h a;
#pragma unroll
  for (int h = 0; h < 16; ++h) {
    int k = kb + (h & 7) + ((h >> 3) << 4);
    a[h] = (_Float16)row[k];
  }
  return a;
}

__device__ __forceinline__ v16h frag_a_lds16(const _Float16* H, int lane) {
  int m  = lane & 15;
  int kb = (lane >> 4) << 3;
  const _Float16* row = H + m * 32;
  v16h a;
#pragma unroll
  for (int h = 0; h < 16; ++h) {
    int k = kb + (h & 7) + ((h >> 3) << 4);
    a[h] = row[k];
  }
  return a;
}

// ---------------- epilogue helper ------------------------------------------
__device__ __forceinline__ void bn_coeffs(const float* bias, const float* bn,
                                          int n, int N, float& bi, float& sc, float& sh) {
  bi = bias ? bias[n] : 0.0f;
  sc = 1.0f; sh = 0.0f;
  if (bn) {
    float g = bn[n], bb = bn[N + n], mm = bn[2 * N + n], vv = bn[3 * N + n];
    sc = g * rsqrtf(vv + EPS_);
    sh = bb - mm * sc;
  }
}

// ---------------- fast WMMA GEMM: M%16==0, N%16==0, K%64==0 ----------------
// out = act(bn(A @ W^T + bias)); A:(M,K), W:(N,K), out:(M,N), all f32.
__global__ __launch_bounds__(32) void k_gemm_fast(
    const float* __restrict__ A, const float* __restrict__ W,
    const float* __restrict__ bias, const float* __restrict__ bn,
    float* __restrict__ out, int M, int N, int K, int do_relu) {
  int lane = threadIdx.x;
  int n0 = blockIdx.x * 16;
  int m0 = blockIdx.y * 16;
  v8f acc0 = {}, acc1 = {};
  for (int k0 = 0; k0 < K; k0 += 64) {
    __builtin_prefetch(W + (long)(n0 + (lane & 15)) * K + k0 + 64, 0, 1);
    v16h a0 = frag_a_fast(A, K, m0, k0,      lane);
    v16h b0 = frag_b_fast(W, K, n0, k0,      lane);
    v16h a1 = frag_a_fast(A, K, m0, k0 + 32, lane);
    v16h b1 = frag_b_fast(W, K, n0, k0 + 32, lane);
    acc0 = WMMA(a0, b0, acc0);          // two independent WMMA chains
    acc1 = WMMA(a1, b1, acc1);
  }
  int n = n0 + (lane & 15);
  float bi, sc, sh;
  bn_coeffs(bias, bn, n, N, bi, sc, sh);
  int mg = (lane >> 4) << 3;
#pragma unroll
  for (int i = 0; i < 8; ++i) {
    float v = (acc0[i] + acc1[i] + bi) * sc + sh;
    if (do_relu) v = fmaxf(v, 0.0f);
    out[(long)(m0 + i + mg) * N + n] = v;
  }
}

// ---------------- edge WMMA GEMM: arbitrary M, N, K ------------------------
__global__ __launch_bounds__(32) void k_gemm_edge(
    const float* __restrict__ A, const float* __restrict__ W,
    const float* __restrict__ bias, const float* __restrict__ bn,
    float* __restrict__ out, int M, int N, int K, int do_relu) {
  int lane = threadIdx.x;
  int n0 = blockIdx.x * 16;
  int m0 = blockIdx.y * 16;
  v8f acc = {};
  for (int k0 = 0; k0 < K; k0 += 32) {
    v16h a = frag_a_edge(A, K, m0, k0, M, K, lane);
    v16h b = frag_b_edge(W, K, n0, k0, N, K, lane);
    acc = WMMA(a, b, acc);
  }
  int n = n0 + (lane & 15);
  if (n >= N) return;
  float bi, sc, sh;
  bn_coeffs(bias, bn, n, N, bi, sc, sh);
  int mg = (lane >> 4) << 3;
#pragma unroll
  for (int i = 0; i < 8; ++i) {
    int m = m0 + i + mg;
    if (m < M) {
      float v = (acc[i] + bi) * sc + sh;
      if (do_relu) v = fmaxf(v, 0.0f);
      out[(long)m * N + n] = v;
    }
  }
}

// ---------------- fused FFN: out = relu(x @ w1^T + b1) @ w2^T + b2 ---------
// x:(M,64), w1:(2048,64), w2:(64,2048). One wave per 16-row tile; the
// 16x2048 hidden activation lives in registers/LDS only. X tile staged into
// LDS by the Tensor Data Mover.
__global__ __launch_bounds__(32) void k_ffn(
    const float* __restrict__ x,
    const float* __restrict__ w1, const float* __restrict__ b1,
    const float* __restrict__ w2, const float* __restrict__ b2,
    float* __restrict__ out) {
  __shared__ float    Xs[16 * 64];     // 4 KB
  __shared__ _Float16 Hs[16 * 32];     // 1 KB
  int lane = threadIdx.x;
  int m0 = blockIdx.x * 16;
  const float* xtile = x + (long)m0 * 64;

#if __has_builtin(__builtin_amdgcn_tensor_load_to_lds)
  {
    // D# group0: count=1 | lds_addr | global_addr | type=2
    unsigned lds_off = (unsigned)(uintptr_t)(void*)Xs;
    unsigned long long ga = (unsigned long long)(uintptr_t)xtile;
    u32x4 g0 = { 1u, lds_off, (unsigned)(ga & 0xffffffffu),
                 (unsigned)(((ga >> 32) & 0x1ffffffu) | 0x80000000u) };
    // group1: data_size=4B(code 2); tensor_dim0=64, tensor_dim1=16;
    // tile_dim0=64, tile_dim1=16; tensor_dim0_stride=64.
    i32x8 g1 = { (int)(2u << 16), (int)(64u << 16), (int)(16u << 16),
                 (int)(64u << 16), 16, 64, 0, 0 };
    i32x4 gz4 = { 0, 0, 0, 0 };
    i32x8 gz8 = { 0, 0, 0, 0, 0, 0, 0, 0 };
    __builtin_amdgcn_tensor_load_to_lds(g0, g1, gz4, gz4, gz8, 0);
    __builtin_amdgcn_s_wait_tensorcnt(0);
  }
#else
  for (int e = lane; e < 16 * 64; e += 32) Xs[e] = xtile[e];
#endif
  __syncthreads();

  v16h a0 = frag_a_lds32(Xs, 64, 0, lane);
  v16h a1 = frag_a_lds32(Xs, 64, 32, lane);
  v8f acc[4] = {{}, {}, {}, {}};
  int nc = lane & 15, mg = (lane >> 4) << 3;

  for (int c0 = 0; c0 < 2048; c0 += 32) {
    __builtin_prefetch(w1 + (long)(c0 + 32 + nc) * 64, 0, 1);
    // stage 1: hidden block 16x32 = relu(X @ w1[c0:c0+32]^T + b1)
    v8f h0 = {}, h1 = {};
    h0 = WMMA(a0, frag_b_fast(w1, 64, c0,      0,  lane), h0);
    h0 = WMMA(a1, frag_b_fast(w1, 64, c0,      32, lane), h0);
    h1 = WMMA(a0, frag_b_fast(w1, 64, c0 + 16, 0,  lane), h1);
    h1 = WMMA(a1, frag_b_fast(w1, 64, c0 + 16, 32, lane), h1);
    float bi0 = b1[c0 + nc], bi1 = b1[c0 + 16 + nc];
#pragma unroll
    for (int i = 0; i < 8; ++i) {
      int m = i + mg;
      Hs[m * 32 + nc]      = (_Float16)fmaxf(h0[i] + bi0, 0.0f);
      Hs[m * 32 + 16 + nc] = (_Float16)fmaxf(h1[i] + bi1, 0.0f);
    }
    __syncthreads();
    // stage 2: acc += Hblock(16x32) @ w2[:, c0:c0+32]^T
    v16h ah = frag_a_lds16(Hs, lane);
#pragma unroll
    for (int j = 0; j < 4; ++j)
      acc[j] = WMMA(ah, frag_b_fast(w2, 2048, j * 16, c0, lane), acc[j]);
    __syncthreads();
  }

#pragma unroll
  for (int j = 0; j < 4; ++j) {
    int n = j * 16 + nc;
    float bi = b2[n];
#pragma unroll
    for (int i = 0; i < 8; ++i)
      out[(long)(m0 + i + mg) * 64 + n] = acc[j][i] + bi;
  }
}

// ---------------- transpose sortvec (B,64,2048) -> x (B*2048, 64) ----------
__global__ void k_transpose(const float* __restrict__ sv, float* __restrict__ x) {
  long i = (long)blockIdx.x * 256 + threadIdx.x;
  if (i >= (long)ROWS_ * F_) return;
  int  f  = (int)(i & 63);
  long bn = i >> 6;
  int  n  = (int)(bn & (N_ - 1));
  int  b  = (int)(bn >> 11);
  x[i] = sv[((long)b * F_ + f) * N_ + n];
}

// ---------------- attention: online softmax, f16 K/V in LDS ----------------
// qkv: (B*N, 192) rows [q|k|v]. out: (B*N, 64). grid (8, NHEAD, B), block 256.
__global__ __launch_bounds__(256) void k_attn(const float* __restrict__ qkv,
                                              float* __restrict__ out) {
  __shared__ _Float16 Ks[1024 * 8];   // 16 KB
  __shared__ _Float16 Vs[1024 * 8];   // 16 KB
  int tid = threadIdx.x;
  int h = blockIdx.y, b = blockIdx.z;
  int n = blockIdx.x * 256 + tid;
  const float scale = 0.35355339059327373f;  // 1/sqrt(8)

  float q[HD_];
  const float* qp = qkv + ((long)b * N_ + n) * 192 + h * HD_;
#pragma unroll
  for (int d = 0; d < HD_; ++d) q[d] = qp[d];

  float mrun = -1e30f, lrun = 0.0f, o[HD_];
#pragma unroll
  for (int d = 0; d < HD_; ++d) o[d] = 0.0f;

  for (int ch = 0; ch < 2; ++ch) {
    for (int e = tid; e < 1024 * 8; e += 256) {
      int j = e >> 3, d = e & 7;
      long row = ((long)b * N_ + ch * 1024 + j) * 192;
      Ks[e] = (_Float16)qkv[row + 64  + h * HD_ + d];
      Vs[e] = (_Float16)qkv[row + 128 + h * HD_ + d];
    }
    __syncthreads();
    for (int j = 0; j < 1024; ++j) {
      const _Float16* kr = &Ks[j * 8];
      float s = 0.0f;
#pragma unroll
      for (int d = 0; d < HD_; ++d) s += q[d] * (float)kr[d];
      s *= scale;
      float mnew = fmaxf(mrun, s);
      float corr = __expf(mrun - mnew);
      float p    = __expf(s - mnew);
      lrun = lrun * corr + p;
      const _Float16* vr = &Vs[j * 8];
#pragma unroll
      for (int d = 0; d < HD_; ++d) o[d] = o[d] * corr + p * (float)vr[d];
      mrun = mnew;
    }
    __syncthreads();
  }
  float inv = 1.0f / lrun;
  float* op = out + ((long)b * N_ + n) * F_ + h * HD_;
#pragma unroll
  for (int d = 0; d < HD_; ++d) op[d] = o[d] * inv;
}

// ---------------- layernorm with residual: x = LN(x + f) -------------------
__global__ __launch_bounds__(256) void k_ln_add(float* __restrict__ x,
                                                const float* __restrict__ f,
                                                const float* __restrict__ g,
                                                const float* __restrict__ bb,
                                                int rows) {
  int wid = threadIdx.x >> 5, lane = threadIdx.x & 31;
  long row = (long)blockIdx.x * 8 + wid;
  if (row >= rows) return;
  float* xr = x + row * 64;
  const float* fr = f + row * 64;
  float t0 = xr[lane] + fr[lane];
  float t1 = xr[lane + 32] + fr[lane + 32];
  float s = t0 + t1;
#pragma unroll
  for (int off = 16; off; off >>= 1) s += __shfl_xor(s, off, 32);
  float mu = s * (1.0f / 64.0f);
  float d0 = t0 - mu, d1 = t1 - mu;
  float vs = d0 * d0 + d1 * d1;
#pragma unroll
  for (int off = 16; off; off >>= 1) vs += __shfl_xor(vs, off, 32);
  float inv = rsqrtf(vs * (1.0f / 64.0f) + EPS_);
  xr[lane]      = d0 * inv * g[lane] + bb[lane];
  xr[lane + 32] = d1 * inv * g[lane + 32] + bb[lane + 32];
}

// ---------------- top-k (iterative argmax, jax tie-break: lowest index) ----
__global__ __launch_bounds__(256) void k_topk(float* __restrict__ scores,
                                              int* __restrict__ idx_out,
                                              float* __restrict__ idx_f) {
  __shared__ float sv[256];
  __shared__ int   si[256];
  int b = blockIdx.x, tid = threadIdx.x;
  float* sc = scores + (long)b * N_;
  for (int t = 0; t < TOPK_; ++t) {
    float best = -1e30f;
    int bi = 0x7fffffff;
    for (int j = tid; j < N_; j += 256) {
      float v = sc[j];
      if (v > best || (v == best && j < bi)) { best = v; bi = j; }
    }
    sv[tid] = best; si[tid] = bi;
    __syncthreads();
    for (int s = 128; s; s >>= 1) {
      if (tid < s) {
        if (sv[tid + s] > sv[tid] ||
            (sv[tid + s] == sv[tid] && si[tid + s] < si[tid])) {
          sv[tid] = sv[tid + s]; si[tid] = si[tid + s];
        }
      }
      __syncthreads();
    }
    if (tid == 0) {
      int w = si[0];
      idx_out[b * TOPK_ + t] = w;
      idx_f[b * TOPK_ + t] = (float)w;
      sc[w] = -1e30f;
    }
    __syncthreads();
  }
}

// ---------------- ball query + grouping ------------------------------------
__global__ void k_group(const float* __restrict__ input,
                        const int* __restrict__ idx,
                        float* __restrict__ grouped) {
  int b = blockIdx.x, q = threadIdx.x;
  if (q >= TOPK_) return;
  const float* inb = input + (long)b * 6 * N_;
  int qi = idx[b * TOPK_ + q];
  float qp[6];
#pragma unroll
  for (int d = 0; d < 6; ++d) qp[d] = inb[d * N_ + qi];
  float* g = grouped + ((long)b * TOPK_ + q) * 390;
#pragma unroll
  for (int d = 0; d < 6; ++d) g[d] = qp[d];
  int cnt = 0, first = qi;
  for (int n = 0; n < N_ && cnt < NSAMP_; ++n) {
    float dx = inb[n] - qp[0];
    float dy = inb[N_ + n] - qp[1];
    float dz = inb[2 * N_ + n] - qp[2];
    if (dx * dx + dy * dy + dz * dz <= R2_) {
      if (cnt == 0) first = n;
      for (int d = 0; d < 6; ++d) g[(1 + cnt) * 6 + d] = inb[d * N_ + n];
      ++cnt;
    }
  }
  float fp[6];
#pragma unroll
  for (int d = 0; d < 6; ++d) fp[d] = inb[d * N_ + first];
  for (; cnt < NSAMP_; ++cnt)
    for (int d = 0; d < 6; ++d) g[(1 + cnt) * 6 + d] = fp[d];
}

// ---------------- repack r (b,k,c) -> h (b, c*16+k) ------------------------
__global__ void k_repack(const float* __restrict__ r2, float* __restrict__ hlin) {
  int i = blockIdx.x * 256 + threadIdx.x;
  if (i >= B_ * 8192) return;
  int b = i >> 13, rem = i & 8191, c = rem >> 4, k = rem & 15;
  hlin[i] = r2[((long)b * TOPK_ + k) * 512 + c];
}

// ---------------------------------------------------------------------------
extern "C" void kernel_launch(void* const* d_in, const int* in_sizes, int n_in,
                              void* d_out, int out_size, void* d_ws, size_t ws_size,
                              hipStream_t stream) {
  (void)in_sizes; (void)n_in; (void)out_size; (void)ws_size;
  const float* sortvec = (const float*)d_in[0];
  const float* input   = (const float*)d_in[1];
  const float* tl_win  = (const float*)d_in[2];
  const float* tl_bin  = (const float*)d_in[3];
  const float* tl_wout = (const float*)d_in[4];
  const float* tl_bout = (const float*)d_in[5];
  const float* tl_w1   = (const float*)d_in[6];
  const float* tl_b1   = (const float*)d_in[7];
  const float* tl_w2   = (const float*)d_in[8];
  const float* tl_b2   = (const float*)d_in[9];
  const float* tl_ln1g = (const float*)d_in[10];
  const float* tl_ln1b = (const float*)d_in[11];
  const float* tl_ln2g = (const float*)d_in[12];
  const float* tl_ln2b = (const float*)d_in[13];
  const float* fg_w0 = (const float*)d_in[14];
  const float* fg_b0 = (const float*)d_in[15];
  const float* fg_w1 = (const float*)d_in[16];
  const float* fg_b1 = (const float*)d_in[17];
  const float* fg_w2 = (const float*)d_in[18];
  const float* fg_b2 = (const float*)d_in[19];
  const float* fbn0  = (const float*)d_in[20];
  const float* fbn1  = (const float*)d_in[21];
  const float* fbn2  = (const float*)d_in[22];
  const float* rc_w0 = (const float*)d_in[23];
  const float* rc_b0 = (const float*)d_in[24];
  const float* rc_w1 = (const float*)d_in[25];
  const float* rc_b1 = (const float*)d_in[26];
  const float* rc_w2 = (const float*)d_in[27];
  const float* rc_b2 = (const float*)d_in[28];
  const float* rbn0  = (const float*)d_in[29];
  const float* rbn1  = (const float*)d_in[30];
  const float* rbn2  = (const float*)d_in[31];
  const float* fl_w0 = (const float*)d_in[32];
  const float* fl_b0 = (const float*)d_in[33];
  const float* fl_w1 = (const float*)d_in[34];
  const float* fl_b1 = (const float*)d_in[35];
  const float* lbn0  = (const float*)d_in[36];
  const float* lbn1  = (const float*)d_in[37];

  // workspace layout (floats). Peak: 3*XN + 3.1M ~ 25 MB.
  const long XN = (long)ROWS_ * F_;  // 1,048,576
  float* ws   = (float*)d_ws;
  float* x    = ws;
  float* bufA = ws + XN;
  float* bufB = ws + 2 * XN;
  float* qkv  = ws + 3 * XN;              // 16384*192 floats
  // post-transformer reuse
  float* s0      = bufA;
  float* s1      = bufB;
  float* scores  = qkv;                   // 16384
  float* grouped = qkv + 16384;           // 128*390
  float* r0      = grouped + 128 * 390;   // 128*128
  float* r1      = r0 + 128 * 128;        // 128*256
  float* r2      = r1 + 128 * 256;        // 128*512
  float* hlin    = r2 + 128 * 512;        // 8*8192
  float* h0      = hlin + 8 * 8192;       // 8*1024
  int*   idxbuf  = (int*)(h0 + 8 * 1024);
  float* outf    = (float*)d_out;

  k_transpose<<<(ROWS_ * F_ + 255) / 256, 256, 0, stream>>>(sortvec, x);

  for (int l = 0; l < 2; ++l) {
    k_gemm_fast<<<dim3(192 / 16, ROWS_ / 16), 32, 0, stream>>>(
        x, tl_win + (long)l * 192 * 64, tl_bin + l * 192, nullptr,
        qkv, ROWS_, 192, 64, 0);
    k_attn<<<dim3(N_ / 256, NHEAD_, B_), 256, 0, stream>>>(qkv, bufA);
    k_gemm_fast<<<dim3(64 / 16, ROWS_ / 16), 32, 0, stream>>>(
        bufA, tl_wout + (long)l * 64 * 64, tl_bout + l * 64, nullptr,
        bufB, ROWS_, 64, 64, 0);
    k_ln_add<<<ROWS_ / 8, 256, 0, stream>>>(x, bufB, tl_ln1g + l * 64,
                                            tl_ln1b + l * 64, ROWS_);
    k_ffn<<<ROWS_ / 16, 32, 0, stream>>>(
        x, tl_w1 + (long)l * 2048 * 64, tl_b1 + l * 2048,
        tl_w2 + (long)l * 64 * 2048, tl_b2 + l * 64, bufB);
    k_ln_add<<<ROWS_ / 8, 256, 0, stream>>>(x, bufB, tl_ln2g + l * 64,
                                            tl_ln2b + l * 64, ROWS_);
  }

  // foreground scoring head
  k_gemm_fast<<<dim3(4, ROWS_ / 16), 32, 0, stream>>>(x,  fg_w0, fg_b0, fbn0, s0,
                                                      ROWS_, 64, 64, 1);
  k_gemm_fast<<<dim3(1, ROWS_ / 16), 32, 0, stream>>>(s0, fg_w1, fg_b1, fbn1, s1,
                                                      ROWS_, 16, 64, 1);
  k_gemm_edge<<<dim3(1, ROWS_ / 16), 32, 0, stream>>>(s1, fg_w2, fg_b2, fbn2, scores,
                                                      ROWS_, 1, 16, 1);

  k_topk<<<B_, 256, 0, stream>>>(scores, idxbuf, outf + B_ * 512);
  k_group<<<B_, TOPK_, 0, stream>>>(input, idxbuf, grouped);

  // radius-conv chain, rows = B*TOPK = 128
  k_gemm_edge<<<dim3(128 / 16, 8), 32, 0, stream>>>(grouped, rc_w0, rc_b0, rbn0, r0,
                                                    128, 128, 390, 1);
  k_gemm_fast<<<dim3(256 / 16, 8), 32, 0, stream>>>(r0, rc_w1, rc_b1, rbn1, r1,
                                                    128, 256, 128, 1);
  k_gemm_fast<<<dim3(512 / 16, 8), 32, 0, stream>>>(r1, rc_w2, rc_b2, rbn2, r2,
                                                    128, 512, 256, 1);

  k_repack<<<(B_ * 8192 + 255) / 256, 256, 0, stream>>>(r2, hlin);

  k_gemm_edge<<<dim3(1024 / 16, 1), 32, 0, stream>>>(hlin, fl_w0, fl_b0, lbn0, h0,
                                                     B_, 1024, 8192, 1);
  k_gemm_edge<<<dim3(512 / 16, 1), 32, 0, stream>>>(h0, fl_w1, fl_b1, lbn1, outf,
                                                    B_, 512, 1024, 1);
}
